// LinearLayer_20031727469177
// MI455X (gfx1250) — matrix operand order repaired and checked by reading, not compiled, not run on hardware
//
#include <hip/hip_runtime.h>

typedef __attribute__((ext_vector_type(16))) __bf16 v16bf;
typedef __attribute__((ext_vector_type(8)))  float  v8f;

union FragU { unsigned u[8]; v16bf v; };

__device__ __forceinline__ float gelu_exact(float x) {
    return 0.5f * x * (1.0f + erff(x * 0.70710678118654752f));
}

// Load one 32x16 bf16 B fragment (ISA B layout: N = lane%16, K = 8*(lane/16)+s for
// s<8 and 16+8*(lane/16)+(s-8) for s>=8, K-pairs packed per dword) as two b128 loads.
__device__ __forceinline__ void load_bfrag(FragU& b, const __bf16* bcol, int ks) {
    const uint4 lo = *(const uint4*)(bcol + ks * 32);
    const uint4 hi = *(const uint4*)(bcol + ks * 32 + 16);
    b.u[0] = lo.x; b.u[1] = lo.y; b.u[2] = lo.z; b.u[3] = lo.w;
    b.u[4] = hi.x; b.u[5] = hi.y; b.u[6] = hi.z; b.u[7] = hi.w;
}

// One fused pipeline: out_block = ((bf16(x_block) @ W0) * s -> normact -> @ W1) * s
template<int MUL, int D, int NT, int OFF>
__global__ __launch_bounds__(256, 1)
void pipe_kernel(const float* __restrict__ x,
                 const float* __restrict__ W0,
                 const float* __restrict__ W1,
                 float* __restrict__ out,
                 int Ntot)
{
    constexpr int M1   = NT * D;        // GEMM rows per tile
    constexpr int SA   = MUL + 8;       // padded stride (halves) for A / I buffers
    constexpr int SW   = MUL + 8;       // padded stride (halves) for W^T buffer
    constexpr int MT   = M1 / 16;       // 16-row tiles
    constexpr int NTL  = MUL / 16;      // 16-col tiles
    constexpr int NK   = MUL / 32;      // K steps of 32
    constexpr int CW   = MUL * D;       // block row width in the 960-wide feature
    constexpr int DTOT = 960;

    extern __shared__ char smem[];
    __bf16* Abuf = (__bf16*)smem;            // [M1][SA]   bf16 A tile
    __bf16* Ibuf = Abuf + M1 * SA;           // [M1][SA]   bf16 intermediate
    __bf16* Wbuf = Ibuf + M1 * SA;           // [MUL][SW]  bf16 W^T: Wbuf[w][u]

    const int tid  = threadIdx.x;
    const int wave = tid >> 5;
    const int lane = tid & 31;
    const int lrow = lane & 15;   // N col within C/B frags, M row within A frag
    const int grp  = lane >> 4;   // lane-half selector
    const int n0   = blockIdx.x * NT;
    const float scale = 1.0f / sqrtf((float)MUL);

    // ---- Phase 0: stage x tile (de-interleaved, b128 reads) + W0^T into LDS ----
    for (int idx4 = tid; idx4 < NT * CW / 4; idx4 += 256) {
        int base = idx4 * 4;
        int nl = base / CW;
        int c0 = base - nl * CW;
        float4 v = make_float4(0.f, 0.f, 0.f, 0.f);
        if (n0 + nl < Ntot)
            v = *(const float4*)(x + (size_t)(n0 + nl) * DTOT + OFF + c0);
        const float f[4] = {v.x, v.y, v.z, v.w};
#pragma unroll
        for (int j = 0; j < 4; ++j) {
            int c = c0 + j;
            int u = c / D;
            int i = c - u * D;
            Abuf[(nl * D + i) * SA + u] = (__bf16)f[j];
        }
    }
    for (int idx4 = tid; idx4 < MUL * MUL / 4; idx4 += 256) {
        int base = idx4 * 4;
        int u  = base / MUL;
        int w0 = base - u * MUL;
        float4 v = *(const float4*)(W0 + base);
        Wbuf[(w0 + 0) * SW + u] = (__bf16)v.x;
        Wbuf[(w0 + 1) * SW + u] = (__bf16)v.y;
        Wbuf[(w0 + 2) * SW + u] = (__bf16)v.z;
        Wbuf[(w0 + 3) * SW + u] = (__bf16)v.w;
    }
    __syncthreads();

    // ---- Phase 1: GEMM1, Ibuf = bf16((Abuf @ W0) * scale) ----
    for (int mt = wave; mt < MT; mt += 8) {
        // A panel for this 16-row stripe, all K steps, b128 loads
        FragU areg[NK];
        const __bf16* arow = Abuf + (mt * 16 + lrow) * SA + (grp << 3);
#pragma unroll
        for (int ks = 0; ks < NK; ++ks) {
            const uint4 lo = *(const uint4*)(arow + ks * 32);
            const uint4 hi = *(const uint4*)(arow + ks * 32 + 16);
            areg[ks].u[0] = lo.x; areg[ks].u[1] = lo.y;
            areg[ks].u[2] = lo.z; areg[ks].u[3] = lo.w;
            areg[ks].u[4] = hi.x; areg[ks].u[5] = hi.y;
            areg[ks].u[6] = hi.z; areg[ks].u[7] = hi.w;
        }

        for (int nt = 0; nt < NTL; ++nt) {
            v8f acc = {};
            const __bf16* bcol = Wbuf + (nt * 16 + lrow) * SW + (grp << 3);
            FragU b[2];
            load_bfrag(b[0], bcol, 0);
#pragma unroll
            for (int ks = 0; ks < NK; ++ks) {
                if (ks + 1 < NK) load_bfrag(b[(ks + 1) & 1], bcol, ks + 1);
                acc = __builtin_amdgcn_wmma_f32_16x16x32_bf16(
                        false, areg[ks].v, false, b[ks & 1].v, (short)0, acc, false, false);
            }
#pragma unroll
            for (int r = 0; r < 8; ++r) {
                int m = mt * 16 + r + (grp << 3);
                Ibuf[m * SA + nt * 16 + lrow] = (__bf16)(acc[r] * scale);
            }
        }
    }
    __syncthreads();

    // ---- Phase 2: load W1^T (reuse Wbuf) + norm-activation in place ----
    for (int idx4 = tid; idx4 < MUL * MUL / 4; idx4 += 256) {
        int base = idx4 * 4;
        int u  = base / MUL;
        int w0 = base - u * MUL;
        float4 v = *(const float4*)(W1 + base);
        Wbuf[(w0 + 0) * SW + u] = (__bf16)v.x;
        Wbuf[(w0 + 1) * SW + u] = (__bf16)v.y;
        Wbuf[(w0 + 2) * SW + u] = (__bf16)v.z;
        Wbuf[(w0 + 3) * SW + u] = (__bf16)v.w;
    }
    for (int g = tid; g < NT * MUL; g += 256) {
        int nl = g / MUL;
        int w  = g - nl * MUL;
        __bf16* p = Ibuf + (nl * D) * SA + w;
        float comp[D];
        float ss = 0.0f;
#pragma unroll
        for (int i = 0; i < D; ++i) {
            comp[i] = (float)p[i * SA];
            ss += comp[i] * comp[i];
        }
        float nrm = sqrtf(ss);
        float s = (nrm > 0.0f) ? (gelu_exact(nrm) / nrm) : 0.0f;
#pragma unroll
        for (int i = 0; i < D; ++i)
            p[i * SA] = (__bf16)(comp[i] * s);
    }
    __syncthreads();

    // ---- Phase 3: GEMM2, out = (Ibuf @ W1) * scale, scatter to interleaved layout ----
    for (int mt = wave; mt < MT; mt += 8) {
        FragU areg[NK];
        const __bf16* arow = Ibuf + (mt * 16 + lrow) * SA + (grp << 3);
#pragma unroll
        for (int ks = 0; ks < NK; ++ks) {
            const uint4 lo = *(const uint4*)(arow + ks * 32);
            const uint4 hi = *(const uint4*)(arow + ks * 32 + 16);
            areg[ks].u[0] = lo.x; areg[ks].u[1] = lo.y;
            areg[ks].u[2] = lo.z; areg[ks].u[3] = lo.w;
            areg[ks].u[4] = hi.x; areg[ks].u[5] = hi.y;
            areg[ks].u[6] = hi.z; areg[ks].u[7] = hi.w;
        }

        for (int nt = 0; nt < NTL; ++nt) {
            v8f acc = {};
            const __bf16* bcol = Wbuf + (nt * 16 + lrow) * SW + (grp << 3);
            FragU b[2];
            load_bfrag(b[0], bcol, 0);
#pragma unroll
            for (int ks = 0; ks < NK; ++ks) {
                if (ks + 1 < NK) load_bfrag(b[(ks + 1) & 1], bcol, ks + 1);
                acc = __builtin_amdgcn_wmma_f32_16x16x32_bf16(
                        false, areg[ks].v, false, b[ks & 1].v, (short)0, acc, false, false);
            }
#pragma unroll
            for (int r = 0; r < 8; ++r) {
                int m  = mt * 16 + r + (grp << 3);
                int ng = n0 + m / D;
                int i  = m - (m / D) * D;
                if (ng < Ntot)
                    out[(size_t)ng * DTOT + OFF + (nt * 16 + lrow) * D + i] = acc[r] * scale;
            }
        }
    }
}

extern "C" void kernel_launch(void* const* d_in, const int* in_sizes, int n_in,
                              void* d_out, int out_size, void* d_ws, size_t ws_size,
                              hipStream_t stream) {
    const float* x   = (const float*)d_in[0];
    const float* W00 = (const float*)d_in[1];
    const float* W01 = (const float*)d_in[2];
    const float* W02 = (const float*)d_in[3];
    const float* W10 = (const float*)d_in[4];
    const float* W11 = (const float*)d_in[5];
    const float* W12 = (const float*)d_in[6];
    float* out = (float*)d_out;

    const int Ntot = in_sizes[0] / 960;

    // LDS bytes: (2*M1*SA + MUL*SW) * 2
    const int smem0 = (2 * 128 * 264 + 256 * 264) * 2; // 270336
    const int smem1 = (2 * 192 * 136 + 128 * 136) * 2; // 139264
    const int smem2 = (2 * 320 * 72  + 64  * 72 ) * 2; // 101376

    (void)hipFuncSetAttribute((const void*)pipe_kernel<256, 1, 128, 0>,
                              hipFuncAttributeMaxDynamicSharedMemorySize, smem0);
    (void)hipFuncSetAttribute((const void*)pipe_kernel<128, 3, 64, 256>,
                              hipFuncAttributeMaxDynamicSharedMemorySize, smem1);
    (void)hipFuncSetAttribute((const void*)pipe_kernel<64, 5, 64, 640>,
                              hipFuncAttributeMaxDynamicSharedMemorySize, smem2);

    pipe_kernel<256, 1, 128, 0><<<(Ntot + 127) / 128, 256, smem0, stream>>>(x, W00, W10, out, Ntot);
    pipe_kernel<128, 3, 64, 256><<<(Ntot + 63) / 64, 256, smem1, stream>>>(x, W01, W11, out, Ntot);
    pipe_kernel<64, 5, 64, 640><<<(Ntot + 63) / 64, 256, smem2, stream>>>(x, W02, W12, out, Ntot);
}
